// TransformerEncoderReadout_31808527794437
// MI455X (gfx1250) — matrix operand order repaired and checked by reading, compiled, not run on hardware
//
#include <hip/hip_runtime.h>
#include <hip/hip_bf16.h>

// ---------------- problem constants (match reference) ----------------
#define BB    64
#define MAXN  512
#define DD    256
#define HH    8
#define KD    256          // per-head key dim
#define FF    1024
#define NTOT  25600
#define HK    (HH*KD)      // 2048
#define LN_EPS 1e-3f

typedef __attribute__((ext_vector_type(16))) _Float16 v16h;
typedef __attribute__((ext_vector_type(8)))  float    v8f;
typedef unsigned int u32;
typedef u32 u32x4 __attribute__((ext_vector_type(4)));
typedef u32 u32x8 __attribute__((ext_vector_type(8)));

// LDS stage: 64 rows x (64 data + 24 pad) halves; 176B row stride is 16B-aligned
// and maps the 16 rows of a B-fragment ds_load_b128 onto distinct bank groups.
#define BSTRIDE 88

// ---------------- WMMA fragment loaders (wave32; contiguous k = 16*(lane/16) + j for A and B) ------
// A matrix 16x32 f16: lane L (0-15) row M=L, halves j0..15 -> K=0..15
//                     lane L (16-31) row M=L-16, K=16..31
__device__ __forceinline__ v16h load_a_frag(const _Float16* p, int lda) {
  int lane = threadIdx.x & 31;
  int m = lane & 15;
  int kb = (lane >> 4) << 4;                  // 0 or 16
  const _Float16* r = p + (size_t)m * lda;
  union { uint4 u[2]; v16h v; } u;
  u.u[0] = *(const uint4*)(r + kb);           // K = kb..kb+7
  u.u[1] = *(const uint4*)(r + kb + 8);       // K = kb+8..kb+15
  return u.v;
}

// B matrix 32x16 f16, stored as B^T row-major [N][K]: lane L col N=L%16,
// halves j -> K = 16*(L/16) + j  (16 contiguous halves)
__device__ __forceinline__ v16h load_bt_frag(const _Float16* bt, int ldbt) {
  int lane = threadIdx.x & 31;
  int n = lane & 15;
  int kb = (lane >> 4) << 4;                  // 0 or 16
  const _Float16* r = bt + (size_t)n * ldbt + kb;
  union { uint4 u[2]; v16h v; } u;
  u.u[0] = *(const uint4*)(r);
  u.u[1] = *(const uint4*)(r + 8);
  return u.v;
}

// Same B-fragment pattern but from the padded LDS stage (ds_load_b128 path)
__device__ __forceinline__ v16h load_bt_frag_lds(const _Float16* bt) {
  int lane = threadIdx.x & 31;
  int n = lane & 15;
  int kb = (lane >> 4) << 4;
  const _Float16* r = bt + n * BSTRIDE + kb;
  union { uint4 u[2]; v16h v; } u;
  u.u[0] = *(const uint4*)(r);
  u.u[1] = *(const uint4*)(r + 8);
  return u.v;
}

__device__ __forceinline__ v8f wmma_f16(v16h a, v16h b, v8f c) {
  return __builtin_amdgcn_wmma_f32_16x16x32_f16(false, a, false, b, (short)0, c,
                                                false, false);
}

// ---------------- Tensor Data Mover: 64x64 f16 tile -> padded LDS -----
// D# per CDNA5 ISA. data_size=2B, tile 64x64,
// pad_enable: pad_interval=32 DWORDs (one 128B row), pad_amount=12 DWORDs (48B)
// -> LDS row stride 176B == BSTRIDE halves.
__device__ __forceinline__ void tdm_load_b_tile(const _Float16* gsrc, unsigned lds_off,
                                                int ldbt_elems, int kd_remaining) {
  unsigned long long ga = (unsigned long long)(uintptr_t)gsrc;
  u32x4 g0;
  g0[0] = 1u;                                              // count=1 (valid user D#)
  g0[1] = lds_off;                                         // lds_addr (bytes)
  g0[2] = (u32)(ga & 0xffffffffu);                         // global_addr[31:0]
  g0[3] = (u32)((ga >> 32) & 0x01ffffffu) | (2u << 30);    // global_addr[56:32], type=2
  u32x8 g1;
  g1[0] = (1u << 16)            // data_size = 2 bytes
        | (1u << 20)            // pad_enable
        | (4u << 22)            // pad_interval: 32 DWORDs
        | (11u << 25);          // pad_amount: 12 DWORDs
  g1[1] = ((u32)kd_remaining & 0xffffu) << 16;             // tensor_dim0[15:0]
  g1[2] = (((u32)kd_remaining >> 16) & 0xffffu)            // tensor_dim0[31:16]
        | (64u << 16);                                     // tensor_dim1[15:0]
  g1[3] = (64u << 16);                                     // tile_dim0 = 64
  g1[4] = 64u;                                             // tile_dim1 = 64, tile_dim2 = 0
  g1[5] = (u32)ldbt_elems;                                 // tensor_dim0_stride[31:0]
  g1[6] = 0u;                                              // stride[47:32], dim1_stride lo
  g1[7] = 0u;
  asm volatile("tensor_load_to_lds %0, %1" :: "s"(g0), "s"(g1) : "memory");
}

// ---------------- partition padding ----------------------------------
__global__ void count_kernel(const int* __restrict__ seg, int* __restrict__ counts) {
  int i = blockIdx.x * blockDim.x + threadIdx.x;
  if (i < NTOT) atomicAdd(&counts[seg[i]], 1);
}

__global__ void starts_kernel(const int* __restrict__ counts, int* __restrict__ starts) {
  if (threadIdx.x == 0) {
    int s = 0;
    for (int b = 0; b < BB; ++b) { starts[b] = s; s += counts[b]; }
  }
}

__global__ void scatter_kernel(const float* __restrict__ feat, const int* __restrict__ seg,
                               const int* __restrict__ starts, float* __restrict__ xf,
                               _Float16* __restrict__ xh, int* __restrict__ mask) {
  __shared__ int nz;
  int i = blockIdx.x;           // node
  int d = threadIdx.x;          // feature
  if (d == 0) nz = 0;
  __syncthreads();
  int b = seg[i];
  int p = i - starts[b];
  float v = feat[(size_t)i * DD + d];
  size_t o = ((size_t)b * MAXN + p) * DD + d;
  xf[o] = v;
  xh[o] = (_Float16)v;
  if (v != 0.0f) atomicOr(&nz, 1);
  __syncthreads();
  if (d == 0) mask[b * MAXN + p] = nz;
}

// ---------------- weight conversion ----------------------------------
__global__ void cvt_qkvw_kernel(const float* __restrict__ Wq, const float* __restrict__ Wk,
                                const float* __restrict__ Wv, _Float16* __restrict__ wqT,
                                _Float16* __restrict__ wkT, _Float16* __restrict__ wvT) {
  int i = blockIdx.x * blockDim.x + threadIdx.x;
  if (i >= DD * HH * KD) return;
  int k = i % KD;
  int h = (i / KD) % HH;
  int d = i / (KD * HH);
  size_t dst = ((size_t)h * KD + k) * DD + d;
  wqT[dst] = (_Float16)Wq[i];
  wkT[dst] = (_Float16)Wk[i];
  wvT[dst] = (_Float16)Wv[i];
}

__global__ void tcvt_kernel(const float* __restrict__ src, _Float16* __restrict__ dst,
                            int R, int C) {
  int i = blockIdx.x * blockDim.x + threadIdx.x;
  if (i >= R * C) return;
  int r = i / C, c = i % C;
  dst[(size_t)c * R + r] = (_Float16)src[i];
}

// ---------------- QKV projection (per b,h WMMA GEMM 512x256x256) ------
__global__ void qkv_kernel(const _Float16* __restrict__ xh,
                           const _Float16* __restrict__ wqT, const _Float16* __restrict__ wkT,
                           const _Float16* __restrict__ wvT,
                           const float* __restrict__ bq, const float* __restrict__ bk,
                           const float* __restrict__ bv,
                           _Float16* __restrict__ qb, _Float16* __restrict__ kb,
                           _Float16* __restrict__ vT) {
  int bh  = blockIdx.y;
  int b   = bh / HH, h = bh % HH;
  int mat = blockIdx.z;                                  // 0=q 1=k 2=v
  int wave = (blockIdx.x * blockDim.x + threadIdx.x) >> 5;   // 0..127
  int tm = wave >> 2;
  int tn = wave & 3;
  int lane = threadIdx.x & 31;

  const _Float16* A  = xh + (size_t)b * MAXN * DD + (size_t)(tm * 16) * DD;
  const _Float16* Bt = (mat == 0 ? wqT : mat == 1 ? wkT : wvT) + (size_t)h * KD * DD;
  const float*  bias = (mat == 0 ? bq  : mat == 1 ? bk  : bv ) + h * KD;

  v8f acc[4] = {};
  for (int k = 0; k < DD; k += 32) {
    v16h af = load_a_frag(A + k, DD);
#pragma unroll
    for (int j = 0; j < 4; ++j) {
      v16h bf = load_bt_frag(Bt + (size_t)(tn * 64 + j * 16) * DD + k, DD);
      acc[j] = wmma_f16(af, bf, acc[j]);
    }
  }
  int c0 = lane & 15, hi = lane >> 4;
  float scale = (mat == 0) ? 0.0625f : 1.0f;             // 1/sqrt(256) folded into q
  if (mat < 2) {
    _Float16* out = (mat == 0 ? qb : kb) + (size_t)bh * MAXN * KD;
#pragma unroll
    for (int j = 0; j < 4; ++j) {
      int col = tn * 64 + j * 16 + c0;
      float bval = bias[col];
#pragma unroll
      for (int r = 0; r < 8; ++r) {
        int row = tm * 16 + r + 8 * hi;
        out[(size_t)row * KD + col] = (_Float16)((acc[j][r] + bval) * scale);
      }
    }
  } else {                                               // v stored transposed [kfeat][node]
    _Float16* out = vT + (size_t)bh * KD * MAXN;
#pragma unroll
    for (int j = 0; j < 4; ++j) {
      int col = tn * 64 + j * 16 + c0;
      float bval = bias[col];
#pragma unroll
      for (int r = 0; r < 8; ++r) {
        int row = tm * 16 + r + 8 * hi;
        out[(size_t)col * MAXN + row] = (_Float16)(acc[j][r] + bval);
      }
    }
  }
}

// ---------------- flash attention (one wave = 16 query rows) ----------
__global__ void attn_kernel(const _Float16* __restrict__ qb, const _Float16* __restrict__ kb,
                            const _Float16* __restrict__ vT, const int* __restrict__ mask,
                            _Float16* __restrict__ ocat) {
  __shared__ __attribute__((aligned(16))) _Float16 pls[8 * 16 * 32];
  int bh = blockIdx.y;
  int b = bh / HH, h = bh % HH;
  int warp = threadIdx.x >> 5;
  int lane = threadIdx.x & 31;
  int q0 = blockIdx.x * 128 + warp * 16;

  const _Float16* qp = qb + (size_t)bh * MAXN * KD + (size_t)q0 * KD;
  const _Float16* kp = kb + (size_t)bh * MAXN * KD;
  const _Float16* vp = vT + (size_t)bh * KD * MAXN;

  v16h qf[8];
#pragma unroll
  for (int kk = 0; kk < 8; ++kk) qf[kk] = load_a_frag(qp + kk * 32, KD);

  float rm[8], rl[8];
#pragma unroll
  for (int r = 0; r < 8; ++r) { rm[r] = -3.0e38f; rl[r] = 0.0f; }
  v8f oacc[16] = {};
  int c0 = lane & 15, hi = lane >> 4;
  _Float16* pbuf = pls + warp * 16 * 32;

  for (int mb = 0; mb < 16; ++mb) {                      // 32 keys per block
    v8f s0 = {}, s1 = {};
#pragma unroll
    for (int kk = 0; kk < 8; ++kk) {
      v16h bf0 = load_bt_frag(kp + (size_t)(mb * 32) * KD + kk * 32, KD);
      v16h bf1 = load_bt_frag(kp + (size_t)(mb * 32 + 16) * KD + kk * 32, KD);
      s0 = wmma_f16(qf[kk], bf0, s0);
      s1 = wmma_f16(qf[kk], bf1, s1);
    }
    float bias0 = mask[b * MAXN + mb * 32 + c0]      ? 0.0f : -1.0e9f;
    float bias1 = mask[b * MAXN + mb * 32 + 16 + c0] ? 0.0f : -1.0e9f;
#pragma unroll
    for (int r = 0; r < 8; ++r) {
      float a  = s0[r] + bias0;
      float bb = s1[r] + bias1;
      float t = fmaxf(a, bb);
#pragma unroll
      for (int off = 1; off < 16; off <<= 1) t = fmaxf(t, __shfl_xor(t, off, 32));
      float nm = fmaxf(rm[r], t);
      float sc = __expf(rm[r] - nm);
      float p0 = __expf(a  - nm);
      float p1 = __expf(bb - nm);
      float ts = p0 + p1;
#pragma unroll
      for (int off = 1; off < 16; off <<= 1) ts += __shfl_xor(ts, off, 32);
      rl[r] = rl[r] * sc + ts;
      rm[r] = nm;
#pragma unroll
      for (int t16 = 0; t16 < 16; ++t16) oacc[t16][r] *= sc;
      int row = r + 8 * hi;                              // accumulator-layout row
      pbuf[row * 32 + c0]      = (_Float16)p0;
      pbuf[row * 32 + 16 + c0] = (_Float16)p1;
    }
    // LDS round-trip converts accumulator layout -> A-fragment layout
    v16h pf = load_a_frag(pbuf, 32);
#pragma unroll
    for (int t16 = 0; t16 < 16; ++t16) {
      v16h vf = load_bt_frag(vp + (size_t)(t16 * 16) * MAXN + mb * 32, MAXN);
      oacc[t16] = wmma_f16(pf, vf, oacc[t16]);
    }
  }

  float inv[8];
#pragma unroll
  for (int r = 0; r < 8; ++r) inv[r] = 1.0f / rl[r];
  _Float16* op = ocat + (size_t)b * MAXN * HK + (size_t)h * KD;
#pragma unroll
  for (int t16 = 0; t16 < 16; ++t16) {
    int col = t16 * 16 + c0;
#pragma unroll
    for (int r = 0; r < 8; ++r) {
      int row = q0 + r + 8 * hi;
      op[(size_t)row * HK + col] = (_Float16)(oacc[t16][r] * inv[r]);
    }
  }
}

// ---------------- generic WMMA GEMM with TDM-staged B -----------------
// Block = 8 waves covering a 128-row x 64-col strip; all waves share the B
// strip, which is DMA'd chunk-by-chunk into padded LDS by the Tensor Data
// Mover (wave 0 issues, TENSORcnt waited, block barrier, ds_load_b128 reads).
__global__ void gemm_bt_kernel(const _Float16* __restrict__ A, int lda,
                               const _Float16* __restrict__ Bt, int ldbt,
                               const float* __restrict__ bias,
                               const float* __restrict__ resid,
                               float* __restrict__ outF, _Float16* __restrict__ outH,
                               int M, int N, int Kd, int relu) {
  __shared__ __attribute__((aligned(16))) _Float16 bstage[64 * BSTRIDE];
  int strips = N >> 6;
  int tm8 = blockIdx.x / strips;
  int tn  = blockIdx.x % strips;
  int warp = threadIdx.x >> 5;
  int lane = threadIdx.x & 31;

  const _Float16* Ap = A + (size_t)(tm8 * 128 + warp * 16) * lda;
  const _Float16* Bp = Bt + (size_t)(tn * 64) * ldbt;
  unsigned lds_off = (unsigned)(uintptr_t)(&bstage[0]);

  v8f acc[4] = {};
  for (int k0 = 0; k0 < Kd; k0 += 64) {
    __syncthreads();                      // previous chunk fully consumed
    if (threadIdx.x < 32) {               // wave 0 drives the TDM
      tdm_load_b_tile(Bp + k0, lds_off, ldbt, Kd - k0);
      __builtin_amdgcn_s_wait_tensorcnt(0);
    }
    __syncthreads();                      // staged B visible to all waves
    if (k0 + 64 < Kd) __builtin_prefetch(Ap + k0 + 64, 0, 3);
#pragma unroll
    for (int kk = 0; kk < 64; kk += 32) {
      v16h af = load_a_frag(Ap + k0 + kk, lda);
#pragma unroll
      for (int j = 0; j < 4; ++j) {
        v16h bf = load_bt_frag_lds(&bstage[(j * 16) * BSTRIDE + kk]);
        acc[j] = wmma_f16(af, bf, acc[j]);
      }
    }
  }

  int c0 = lane & 15, hi = lane >> 4;
#pragma unroll
  for (int j = 0; j < 4; ++j) {
    int col = tn * 64 + j * 16 + c0;
    float bval = bias ? bias[col] : 0.0f;
#pragma unroll
    for (int r = 0; r < 8; ++r) {
      int row = tm8 * 128 + warp * 16 + r + 8 * hi;
      size_t o = (size_t)row * N + col;
      float v = acc[j][r] + bval;
      if (resid) v += resid[o];
      if (relu) v = fmaxf(v, 0.0f);
      if (outF) outF[o] = v;
      if (outH) outH[o] = (_Float16)v;
    }
  }
}

// ---------------- rowwise LayerNorm (wave per 256-wide row) -----------
__global__ void ln_kernel(const float* __restrict__ in, const float* __restrict__ g,
                          const float* __restrict__ bb, float* __restrict__ outF,
                          _Float16* __restrict__ outH) {
  int warp = threadIdx.x >> 5, lane = threadIdx.x & 31;
  int row = blockIdx.x * 8 + warp;
  const float* rp = in + (size_t)row * DD;
  float v[8], s = 0.0f, s2 = 0.0f;
#pragma unroll
  for (int j = 0; j < 8; ++j) {
    v[j] = rp[j * 32 + lane];
    s += v[j];
    s2 += v[j] * v[j];
  }
#pragma unroll
  for (int off = 1; off < 32; off <<= 1) {
    s  += __shfl_xor(s,  off, 32);
    s2 += __shfl_xor(s2, off, 32);
  }
  float mean = s * (1.0f / DD);
  float var  = s2 * (1.0f / DD) - mean * mean;
  float rstd = rsqrtf(var + LN_EPS);
#pragma unroll
  for (int j = 0; j < 8; ++j) {
    int c = j * 32 + lane;
    float o = (v[j] - mean) * rstd * g[c] + bb[c];
    size_t idx = (size_t)row * DD + c;
    if (outF) outF[idx] = o;
    if (outH) outH[idx] = (_Float16)o;
  }
}

// ---------------- global average pool over MAXN -----------------------
__global__ void pool_kernel(const float* __restrict__ in, float* __restrict__ out) {
  int b = blockIdx.x;
  int d = threadIdx.x;
  const float* p = in + (size_t)b * MAXN * DD + d;
  float s = 0.0f;
  for (int n = 0; n < MAXN; ++n) s += p[(size_t)n * DD];
  out[b * DD + d] = s * (1.0f / MAXN);
}

// ---------------- host-side orchestration -----------------------------
extern "C" void kernel_launch(void* const* d_in, const int* in_sizes, int n_in,
                              void* d_out, int out_size, void* d_ws, size_t ws_size,
                              hipStream_t stream) {
  const float* feat  = (const float*)d_in[0];
  const int*   seg   = (const int*)  d_in[1];
  const float* Wq    = (const float*)d_in[2];
  const float* Wk    = (const float*)d_in[3];
  const float* Wv    = (const float*)d_in[4];
  const float* Wo    = (const float*)d_in[5];
  const float* bq    = (const float*)d_in[6];
  const float* bk    = (const float*)d_in[7];
  const float* bv    = (const float*)d_in[8];
  const float* bo    = (const float*)d_in[9];
  const float* ln1g  = (const float*)d_in[10];
  const float* ln1b  = (const float*)d_in[11];
  const float* W1    = (const float*)d_in[12];
  const float* b1    = (const float*)d_in[13];
  const float* W2    = (const float*)d_in[14];
  const float* b2    = (const float*)d_in[15];
  const float* ln2g  = (const float*)d_in[16];
  const float* ln2b  = (const float*)d_in[17];

  char* ws = (char*)d_ws;
  size_t off = 0;
  auto alloc = [&](size_t bytes) -> void* {
    off = (off + 255) & ~(size_t)255;
    void* p = ws + off;
    off += bytes;
    return p;
  };

  int*       counts = (int*)alloc(BB * sizeof(int));
  int*       starts = (int*)alloc(BB * sizeof(int));
  int*       mask   = (int*)alloc((size_t)BB * MAXN * sizeof(int));
  float*     xf     = (float*)alloc((size_t)BB * MAXN * DD * sizeof(float));
  _Float16*  xh     = (_Float16*)alloc((size_t)BB * MAXN * DD * sizeof(_Float16));
  _Float16*  wqT    = (_Float16*)alloc((size_t)HH * KD * DD * sizeof(_Float16));
  _Float16*  wkT    = (_Float16*)alloc((size_t)HH * KD * DD * sizeof(_Float16));
  _Float16*  wvT    = (_Float16*)alloc((size_t)HH * KD * DD * sizeof(_Float16));
  _Float16*  woT    = (_Float16*)alloc((size_t)HK * DD * sizeof(_Float16));
  _Float16*  w1T    = (_Float16*)alloc((size_t)DD * FF * sizeof(_Float16));
  _Float16*  w2T    = (_Float16*)alloc((size_t)FF * DD * sizeof(_Float16));
  _Float16*  qb     = (_Float16*)alloc((size_t)BB * HH * MAXN * KD * sizeof(_Float16));
  _Float16*  kbuf   = (_Float16*)alloc((size_t)BB * HH * MAXN * KD * sizeof(_Float16));
  _Float16*  vT     = (_Float16*)alloc((size_t)BB * HH * KD * MAXN * sizeof(_Float16));
  _Float16*  ocat   = (_Float16*)alloc((size_t)BB * MAXN * HK * sizeof(_Float16));
  float*     y1     = (float*)alloc((size_t)BB * MAXN * DD * sizeof(float));
  float*     hF     = (float*)alloc((size_t)BB * MAXN * DD * sizeof(float));
  _Float16*  hH     = (_Float16*)alloc((size_t)BB * MAXN * DD * sizeof(_Float16));
  // dead-buffer aliases (qb dead after attn; y1 dead after ln1; xf dead after Wo gemm)
  _Float16*  ff1    = qb;
  float*     y2     = y1;
  float*     outFin = xf;

  // zero-init scatter targets
  hipMemsetAsync(counts, 0, BB * sizeof(int), stream);
  hipMemsetAsync(mask,   0, (size_t)BB * MAXN * sizeof(int), stream);
  hipMemsetAsync(xf,     0, (size_t)BB * MAXN * DD * sizeof(float), stream);
  hipMemsetAsync(xh,     0, (size_t)BB * MAXN * DD * sizeof(_Float16), stream);

  // partition padding
  count_kernel<<<(NTOT + 255) / 256, 256, 0, stream>>>(seg, counts);
  starts_kernel<<<1, 32, 0, stream>>>(counts, starts);
  scatter_kernel<<<NTOT, DD, 0, stream>>>(feat, seg, starts, xf, xh, mask);

  // weight conversion
  cvt_qkvw_kernel<<<(DD * HH * KD + 255) / 256, 256, 0, stream>>>(Wq, Wk, Wv, wqT, wkT, wvT);
  tcvt_kernel<<<(HK * DD + 255) / 256, 256, 0, stream>>>(Wo, woT, HK, DD);
  tcvt_kernel<<<(DD * FF + 255) / 256, 256, 0, stream>>>(W1, w1T, DD, FF);
  tcvt_kernel<<<(FF * DD + 255) / 256, 256, 0, stream>>>(W2, w2T, FF, DD);

  // QKV projections: grid (128 waves / 8 per block, B*H, {q,k,v})
  qkv_kernel<<<dim3(16, BB * HH, 3), 256, 0, stream>>>(xh, wqT, wkT, wvT, bq, bk, bv,
                                                       qb, kbuf, vT);

  // attention: 4 blocks of 8 waves cover 512 query rows per (b,h)
  attn_kernel<<<dim3(4, BB * HH), 256, 0, stream>>>(qb, kbuf, vT, mask, ocat);

  // attn_out = ocat @ WoT + bo + x   -> y1
  {
    int blocks = (BB * MAXN / 128) * (DD / 64);          // 1024
    gemm_bt_kernel<<<blocks, 256, 0, stream>>>(ocat, HK, woT, HK, bo, xf,
                                               y1, nullptr, BB * MAXN, DD, HK, 0);
  }
  // h = LN1(y1) -> hF (f32) + hH (f16)
  ln_kernel<<<BB * MAXN / 8, 256, 0, stream>>>(y1, ln1g, ln1b, hF, hH);

  // ff1 = relu(h @ W1T + b1)  (f16 only)
  {
    int blocks = (BB * MAXN / 128) * (FF / 64);          // 4096
    gemm_bt_kernel<<<blocks, 256, 0, stream>>>(hH, DD, w1T, DD, b1, nullptr,
                                               nullptr, ff1, BB * MAXN, FF, DD, 1);
  }
  // y2 = ff1 @ W2T + b2 + h
  {
    int blocks = (BB * MAXN / 128) * (DD / 64);          // 1024
    gemm_bt_kernel<<<blocks, 256, 0, stream>>>(ff1, FF, w2T, FF, b2, hF,
                                               y2, nullptr, BB * MAXN, DD, FF, 0);
  }
  // out = LN2(y2) -> outFin
  ln_kernel<<<BB * MAXN / 8, 256, 0, stream>>>(y2, ln2g, ln2b, outFin, nullptr);

  // global average pooling over the MAXN axis
  pool_kernel<<<BB, DD, 0, stream>>>(outFin, (float*)d_out);
}
